// HierarchicalLSTMDecoder_11957188952339
// MI455X (gfx1250) — matrix operand order repaired and checked by reading, not compiled, not run on hardware
//
#include <hip/hip_runtime.h>
#include <hip/hip_bf16.h>
#include <stdint.h>

// ---------------- problem dims (from reference) ----------------
#define NE_  16
#define SL_  16
#define B_   256
#define IN_  61
#define H_   1024
#define L_   512
#define G_   4096          // 4*H (gates)
#define MD_  4096          // NE*B : decoder chunks batched into one GEMM M-dim
#define INP_ 64            // IN padded to 64 (WMMA N multiple of 16)
#define KT_  64            // target K slice padded 61 -> 64 (K multiple of 32)

typedef __bf16 bf16_t;
typedef __attribute__((ext_vector_type(16))) __bf16 v16bf;
typedef __attribute__((ext_vector_type(8)))  __bf16 v8bf;
typedef __attribute__((ext_vector_type(8)))  float   v8f;

__device__ __forceinline__ v16bf frag_join(v8bf lo, v8bf hi) {
  return __builtin_shufflevector(lo, hi, 0,1,2,3,4,5,6,7,8,9,10,11,12,13,14,15);
}

// A fragment (16x32, bf16), X row-major [*, ld]; ISA layout:
// lanes 0-15: M=lane, K = {k0..k0+7, k0+16..k0+23}; lanes 16-31: M=lane-16, K shifted by 8.
__device__ __forceinline__ v16bf load_fragA(const bf16_t* __restrict__ X, int ld, int m0, int k0) {
  const int lane = threadIdx.x & 31;
  const int half = lane >> 4;
  const int r    = lane & 15;
  const bf16_t* p = X + (size_t)(m0 + r) * ld + (k0 + half * 8);
  v8bf lo = *(const v8bf*)p;          // 16B -> global_load_b128
  v8bf hi = *(const v8bf*)(p + 16);
  return frag_join(lo, hi);
}

// B fragment (32x16, bf16). We compute X @ W^T with W row-major [N, K], so B's column n
// is W's row n (contiguous K). lanes 0-15: N=lane, K = {k0..k0+7, k0+16..k0+23}; lanes 16-31: K shifted by 8.
__device__ __forceinline__ v16bf load_fragB(const bf16_t* __restrict__ W, int ld, int n0, int k0) {
  const int lane = threadIdx.x & 31;
  const int half = lane >> 4;
  const int n    = lane & 15;
  const bf16_t* p = W + (size_t)(n0 + n) * ld + (k0 + half * 8);
  v8bf lo = *(const v8bf*)p;
  v8bf hi = *(const v8bf*)(p + 16);
  return frag_join(lo, hi);
}

__device__ __forceinline__ v8f wmma_bf16(v16bf a, v16bf b, v8f c) {
  // 8 args: (neg_a, A, neg_b, B, c_mod, C, reuse_a, reuse_b)
  return __builtin_amdgcn_wmma_f32_16x16x32_bf16(false, a, false, b, (short)0, c, false, false);
}

__device__ __forceinline__ float fast_sigmoid(float x) { return 1.0f / (1.0f + __expf(-x)); }
__device__ __forceinline__ float fast_tanh(float x) {
  x = fminf(15.0f, fmaxf(-15.0f, x));
  float e = __expf(2.0f * x);
  return (e - 1.0f) / (e + 1.0f);
}

// ---------------- fused LSTM cell ----------------
// z = (zpre[m, 4H] or bias[4H]) + x1 @ w1^T + (x2 ? x2 @ w2^T : 0), gate order i,f,g,o
// c = sig(f)*c + sig(i)*tanh(g); h = sig(o)*tanh(c)  -> h written as bf16 for next GEMM.
// Block = 128 threads (4 waves); each wave owns TWO 16-row M-tiles (B fragments reused
// across both -> 12 b128 loads per 8 WMMAs). blockIdx.y covers 128 rows; blockIdx.x = H-tile.
__global__ __launch_bounds__(128) void lstm_cell_kernel(
    const float* __restrict__ zpre, const float* __restrict__ bias,
    const bf16_t* __restrict__ x1, const bf16_t* __restrict__ w1, int k1,
    const bf16_t* __restrict__ x2, const bf16_t* __restrict__ w2, int k2,
    float* __restrict__ cstate, bf16_t* __restrict__ hout, int M)
{
  const int wave = threadIdx.x >> 5;
  const int m0 = (blockIdx.y * 4 + wave) << 5;   // 2 M-tiles (32 rows) per wave
  const int n0 = blockIdx.x << 4;
  if (m0 >= M) return;                 // wave-uniform; EXEC stays all-1 for WMMA
  const int lane = threadIdx.x & 31;
  const int half = lane >> 4;
  const int ncol = n0 + (lane & 15);   // C/D layout: N = lane%16, M = vgpr + 8*(lane/16)

  v8f acc[2][4];
  if (zpre) {
#pragma unroll
    for (int t2 = 0; t2 < 2; ++t2) {
      const float* zp = zpre + (size_t)(m0 + 16 * t2) * G_ + ncol;
#pragma unroll
      for (int q = 0; q < 4; ++q) {
        const float* zq = zp + q * H_;
#pragma unroll
        for (int e = 0; e < 8; ++e) acc[t2][q][e] = zq[(size_t)(e + 8 * half) * G_];
      }
    }
  } else {
#pragma unroll
    for (int q = 0; q < 4; ++q) {
      float bq = bias[q * H_ + ncol];
#pragma unroll
      for (int e = 0; e < 8; ++e) { acc[0][q][e] = bq; acc[1][q][e] = bq; }
    }
  }

  for (int k0 = 0; k0 < k1; k0 += 32) {
    v16bf a0 = load_fragA(x1, k1, m0,      k0);
    v16bf a1 = load_fragA(x1, k1, m0 + 16, k0);
#pragma unroll
    for (int q = 0; q < 4; ++q) {
      v16bf b = load_fragB(w1, k1, q * H_ + n0, k0);
      acc[0][q] = wmma_bf16(a0, b, acc[0][q]);
      acc[1][q] = wmma_bf16(a1, b, acc[1][q]);
    }
  }
  if (x2) {
    for (int k0 = 0; k0 < k2; k0 += 32) {
      v16bf a0 = load_fragA(x2, k2, m0,      k0);
      v16bf a1 = load_fragA(x2, k2, m0 + 16, k0);
#pragma unroll
      for (int q = 0; q < 4; ++q) {
        v16bf b = load_fragB(w2, k2, q * H_ + n0, k0);
        acc[0][q] = wmma_bf16(a0, b, acc[0][q]);
        acc[1][q] = wmma_bf16(a1, b, acc[1][q]);
      }
    }
  }

#pragma unroll
  for (int t2 = 0; t2 < 2; ++t2) {
#pragma unroll
    for (int e = 0; e < 8; ++e) {
      int m = m0 + 16 * t2 + e + 8 * half;
      size_t off = (size_t)m * H_ + ncol;
      float i = fast_sigmoid(acc[t2][0][e]);
      float f = fast_sigmoid(acc[t2][1][e]);
      float g = fast_tanh(acc[t2][2][e]);
      float o = fast_sigmoid(acc[t2][3][e]);
      float cn = f * cstate[off] + i * g;
      cstate[off] = cn;
      hout[off] = (bf16_t)(o * fast_tanh(cn));
    }
  }
}

// ---------------- generic GEMM: out = act(X @ W^T + bias) ----------------
// Same 2-M-tile register blocking as the cell kernel.
__global__ __launch_bounds__(128) void gemm_bias_kernel(
    const bf16_t* __restrict__ X, const bf16_t* __restrict__ W,
    const float* __restrict__ bias,
    float* __restrict__ outf, bf16_t* __restrict__ outb,
    int M, int N, int K, int act)
{
  const int wave = threadIdx.x >> 5;
  const int m0 = (blockIdx.y * 4 + wave) << 5;
  const int n0 = blockIdx.x << 4;
  if (m0 >= M) return;
  const int lane = threadIdx.x & 31;
  const int half = lane >> 4;
  const int ncol = n0 + (lane & 15);

  v8f acc[2];
  float b0 = bias ? bias[ncol] : 0.0f;
#pragma unroll
  for (int e = 0; e < 8; ++e) { acc[0][e] = b0; acc[1][e] = b0; }

  for (int k0 = 0; k0 < K; k0 += 32) {
    v16bf a0 = load_fragA(X, K, m0,      k0);
    v16bf a1 = load_fragA(X, K, m0 + 16, k0);
    v16bf b = load_fragB(W, K, n0, k0);
    acc[0] = wmma_bf16(a0, b, acc[0]);
    acc[1] = wmma_bf16(a1, b, acc[1]);
  }

#pragma unroll
  for (int t2 = 0; t2 < 2; ++t2) {
#pragma unroll
    for (int e = 0; e < 8; ++e) {
      int m = m0 + 16 * t2 + e + 8 * half;
      float v = acc[t2][e];
      if (act == 1) v = fast_tanh(v);
      size_t off = (size_t)m * N + ncol;
      if (outf) outf[off] = v;
      if (outb) outb[off] = (bf16_t)v;
    }
  }
}

// ---------------- data-prep kernels ----------------
__global__ void cvt_rows_kernel(const float* __restrict__ src, bf16_t* __restrict__ dst,
                                int drows, int srows, int sld, int soff, int ncopy, int dld)
{
  int idx = blockIdx.x * blockDim.x + threadIdx.x;
  if (idx >= drows * dld) return;
  int r = idx / dld, c = idx % dld;
  float v = (r < srows && c < ncopy) ? src[(size_t)r * sld + soff + c] : 0.0f;
  dst[idx] = (bf16_t)v;
}

__global__ void f32_copy_kernel(const float* __restrict__ s, float* __restrict__ d, int n) {
  int i = blockIdx.x * blockDim.x + threadIdx.x;
  if (i < n) d[i] = s[i];
}

__global__ void f32_pad_kernel(const float* __restrict__ s, float* __restrict__ d, int n, int ncopy) {
  int i = blockIdx.x * blockDim.x + threadIdx.x;
  if (i < n) d[i] = (i < ncopy) ? s[i] : 0.0f;
}

__global__ void vec_add_kernel(const float* __restrict__ a, const float* __restrict__ b,
                               float* __restrict__ o, int n) {
  int i = blockIdx.x * blockDim.x + threadIdx.x;
  if (i < n) o[i] = a[i] + b[i];
}

// xt[t][ne*B+b][0..63] = target[(ne*SL+t)*B+b][0..60], zero padded
__global__ void build_xtgt_kernel(const float* __restrict__ target, bf16_t* __restrict__ xt) {
  int idx = blockIdx.x * blockDim.x + threadIdx.x;
  const int total = SL_ * MD_ * KT_;
  if (idx >= total) return;
  int c = idx & (KT_ - 1);
  int mrem = idx >> 6;
  int m = mrem % MD_;
  int t = mrem / MD_;
  int ne = m / B_, b = m % B_;
  float v = 0.0f;
  if (c < IN_) v = target[((size_t)(ne * SL_ + t) * B_ + b) * IN_ + c];
  xt[idx] = (bf16_t)v;
}

// h0_dec layout [NE][2(h/c)][2(layer)][B][H]
__global__ void init_dec_state_kernel(const float* __restrict__ h0dec,
                                      bf16_t* __restrict__ hA, bf16_t* __restrict__ hB,
                                      float* __restrict__ cA, float* __restrict__ cB) {
  int idx = blockIdx.x * blockDim.x + threadIdx.x;
  const int total = NE_ * B_ * H_;
  if (idx >= total) return;
  const size_t S = (size_t)B_ * H_;
  size_t bh = (size_t)idx % S;
  int ne = idx / (B_ * H_);
  size_t base = (size_t)ne * 4 * S + bh;
  hA[idx] = (bf16_t)h0dec[base + 0 * S];
  hB[idx] = (bf16_t)h0dec[base + 1 * S];
  cA[idx] = h0dec[base + 2 * S];
  cB[idx] = h0dec[base + 3 * S];
}

// out[(ne*SL+t)*B+b][0..60] = softmax(logits[m][0..60])
__global__ void softmax_out_kernel(const float* __restrict__ logits, float* __restrict__ out, int t) {
  int m = blockIdx.x * blockDim.x + threadIdx.x;
  if (m >= MD_) return;
  const float* z = logits + (size_t)m * INP_;
  float mx = -3.0e38f;
  for (int c = 0; c < IN_; ++c) mx = fmaxf(mx, z[c]);
  float s = 0.0f;
  for (int c = 0; c < IN_; ++c) s += __expf(z[c] - mx);
  float inv = 1.0f / s;
  int ne = m / B_, b = m % B_;
  float* o = out + ((size_t)(ne * SL_ + t) * B_ + b) * IN_;
  for (int c = 0; c < IN_; ++c) o[c] = __expf(z[c] - mx) * inv;
}

// ---------------- orchestration ----------------
extern "C" void kernel_launch(void* const* d_in, const int* in_sizes, int n_in,
                              void* d_out, int out_size, void* d_ws, size_t ws_size,
                              hipStream_t stream) {
  const float* target = (const float*)d_in[0];
  const float* latent = (const float*)d_in[1];
  const float* h0     = (const float*)d_in[2];
  const float* c0     = (const float*)d_in[3];
  const float* h0dec  = (const float*)d_in[4];
  const float* cW_ih0 = (const float*)d_in[5];
  const float* cW_hh0 = (const float*)d_in[6];
  const float* cb_ih0 = (const float*)d_in[7];
  const float* cb_hh0 = (const float*)d_in[8];
  const float* cW_ih1 = (const float*)d_in[9];
  const float* cW_hh1 = (const float*)d_in[10];
  const float* cb_ih1 = (const float*)d_in[11];
  const float* cb_hh1 = (const float*)d_in[12];
  const float* eW     = (const float*)d_in[13];
  const float* eb     = (const float*)d_in[14];
  const float* dW_ih0 = (const float*)d_in[15];
  const float* dW_hh0 = (const float*)d_in[16];
  const float* db_ih0 = (const float*)d_in[17];
  const float* db_hh0 = (const float*)d_in[18];
  const float* dW_ih1 = (const float*)d_in[19];
  const float* dW_hh1 = (const float*)d_in[20];
  const float* db_ih1 = (const float*)d_in[21];
  const float* db_hh1 = (const float*)d_in[22];
  const float* oW     = (const float*)d_in[23];
  const float* ob     = (const float*)d_in[24];

  // workspace carve (~220 MB, dominated by ZE [4096x4096] f32)
  char* p = (char*)d_ws;
  auto carve = [&](size_t bytes) -> void* {
    void* r = (void*)p;
    p += (bytes + 255) & ~(size_t)255;
    return r;
  };
  float* cb0   = (float*)carve((size_t)G_ * 4);
  float* cb1   = (float*)carve((size_t)G_ * 4);
  float* db0   = (float*)carve((size_t)G_ * 4);
  float* db1   = (float*)carve((size_t)G_ * 4);
  float* obp   = (float*)carve((size_t)INP_ * 4);
  bf16_t* latb   = (bf16_t*)carve((size_t)B_ * L_ * 2);
  bf16_t* Wcih0b = (bf16_t*)carve((size_t)G_ * L_ * 2);
  bf16_t* Wchh0b = (bf16_t*)carve((size_t)G_ * H_ * 2);
  bf16_t* Wcih1b = (bf16_t*)carve((size_t)G_ * H_ * 2);
  bf16_t* Wchh1b = (bf16_t*)carve((size_t)G_ * H_ * 2);
  bf16_t* eWb    = (bf16_t*)carve((size_t)L_ * H_ * 2);
  bf16_t* Wtb    = (bf16_t*)carve((size_t)G_ * KT_ * 2);   // dW_ih0[:, :61] padded
  bf16_t* Web    = (bf16_t*)carve((size_t)G_ * L_ * 2);    // dW_ih0[:, 61:573]
  bf16_t* Wdhh0b = (bf16_t*)carve((size_t)G_ * H_ * 2);
  bf16_t* Wdih1b = (bf16_t*)carve((size_t)G_ * H_ * 2);
  bf16_t* Wdhh1b = (bf16_t*)carve((size_t)G_ * H_ * 2);
  bf16_t* oWb    = (bf16_t*)carve((size_t)INP_ * H_ * 2);  // rows 61..63 zero
  float*  ZX0    = (float*)carve((size_t)B_ * G_ * 4);     // latent@cW_ih0^T + cb0 (step-invariant)
  bf16_t* hcp0   = (bf16_t*)carve((size_t)B_ * H_ * 2);    // conductor layer-A h ping-pong
  bf16_t* hcp1   = (bf16_t*)carve((size_t)B_ * H_ * 2);
  float*  cCondA = (float*)carve((size_t)B_ * H_ * 4);
  float*  cCondB = (float*)carve((size_t)B_ * H_ * 4);
  bf16_t* condH  = (bf16_t*)carve((size_t)(NE_ + 1) * B_ * H_ * 2);  // slot0 = h0[1]
  bf16_t* embb   = (bf16_t*)carve((size_t)MD_ * L_ * 2);
  float*  ZE     = (float*)carve((size_t)MD_ * G_ * 4);    // emb@dW_ih0[:,61:]^T + db0 (step-invariant)
  bf16_t* Xt     = (bf16_t*)carve((size_t)SL_ * MD_ * KT_ * 2);
  bf16_t* hA0    = (bf16_t*)carve((size_t)MD_ * H_ * 2);
  bf16_t* hA1    = (bf16_t*)carve((size_t)MD_ * H_ * 2);
  bf16_t* hB0    = (bf16_t*)carve((size_t)MD_ * H_ * 2);
  bf16_t* hB1    = (bf16_t*)carve((size_t)MD_ * H_ * 2);
  float*  cA     = (float*)carve((size_t)MD_ * H_ * 4);
  float*  cB     = (float*)carve((size_t)MD_ * H_ * 4);
  float*  logits = (float*)carve((size_t)MD_ * INP_ * 4);
  (void)in_sizes; (void)n_in; (void)out_size; (void)ws_size;

  const int TB = 256;
  auto blocks = [&](long long n) { return dim3((unsigned)((n + TB - 1) / TB)); };

  // --- bias sums + pads ---
  vec_add_kernel<<<blocks(G_), TB, 0, stream>>>(cb_ih0, cb_hh0, cb0, G_);
  vec_add_kernel<<<blocks(G_), TB, 0, stream>>>(cb_ih1, cb_hh1, cb1, G_);
  vec_add_kernel<<<blocks(G_), TB, 0, stream>>>(db_ih0, db_hh0, db0, G_);
  vec_add_kernel<<<blocks(G_), TB, 0, stream>>>(db_ih1, db_hh1, db1, G_);
  f32_pad_kernel<<<1, 64, 0, stream>>>(ob, obp, INP_, IN_);

  // --- fp32 -> bf16 conversions (weights read many times; bf16 halves L2 footprint) ---
  cvt_rows_kernel<<<blocks((long long)B_ * L_), TB, 0, stream>>>(latent, latb, B_, B_, L_, 0, L_, L_);
  cvt_rows_kernel<<<blocks((long long)G_ * L_), TB, 0, stream>>>(cW_ih0, Wcih0b, G_, G_, L_, 0, L_, L_);
  cvt_rows_kernel<<<blocks((long long)G_ * H_), TB, 0, stream>>>(cW_hh0, Wchh0b, G_, G_, H_, 0, H_, H_);
  cvt_rows_kernel<<<blocks((long long)G_ * H_), TB, 0, stream>>>(cW_ih1, Wcih1b, G_, G_, H_, 0, H_, H_);
  cvt_rows_kernel<<<blocks((long long)G_ * H_), TB, 0, stream>>>(cW_hh1, Wchh1b, G_, G_, H_, 0, H_, H_);
  cvt_rows_kernel<<<blocks((long long)L_ * H_), TB, 0, stream>>>(eW, eWb, L_, L_, H_, 0, H_, H_);
  cvt_rows_kernel<<<blocks((long long)G_ * KT_), TB, 0, stream>>>(dW_ih0, Wtb, G_, G_, IN_ + L_, 0, IN_, KT_);
  cvt_rows_kernel<<<blocks((long long)G_ * L_), TB, 0, stream>>>(dW_ih0, Web, G_, G_, IN_ + L_, IN_, L_, L_);
  cvt_rows_kernel<<<blocks((long long)G_ * H_), TB, 0, stream>>>(dW_hh0, Wdhh0b, G_, G_, H_, 0, H_, H_);
  cvt_rows_kernel<<<blocks((long long)G_ * H_), TB, 0, stream>>>(dW_ih1, Wdih1b, G_, G_, H_, 0, H_, H_);
  cvt_rows_kernel<<<blocks((long long)G_ * H_), TB, 0, stream>>>(dW_hh1, Wdhh1b, G_, G_, H_, 0, H_, H_);
  cvt_rows_kernel<<<blocks((long long)INP_ * H_), TB, 0, stream>>>(oW, oWb, INP_, IN_, H_, 0, H_, H_);

  // --- conductor init: h states bf16, c states fp32 (copied: never mutate inputs) ---
  cvt_rows_kernel<<<blocks((long long)B_ * H_), TB, 0, stream>>>(h0, hcp0, B_, B_, H_, 0, H_, H_);
  cvt_rows_kernel<<<blocks((long long)B_ * H_), TB, 0, stream>>>(h0 + (size_t)B_ * H_, condH, B_, B_, H_, 0, H_, H_);
  f32_copy_kernel<<<blocks((long long)B_ * H_), TB, 0, stream>>>(c0, cCondA, B_ * H_);
  f32_copy_kernel<<<blocks((long long)B_ * H_), TB, 0, stream>>>(c0 + (size_t)B_ * H_, cCondB, B_ * H_);

  // --- step-invariant conductor input projection: ZX0 = latent @ cW_ih0^T + cb0 ---
  gemm_bias_kernel<<<dim3(G_ / 16, B_ / 128), 128, 0, stream>>>(
      latb, Wcih0b, cb0, ZX0, nullptr, B_, G_, L_, 0);

  // --- conductor: 16 sequential 2-layer LSTM steps ---
  bf16_t* hcp[2] = {hcp0, hcp1};
  for (int s = 0; s < NE_; ++s) {
    bf16_t* hAprev = hcp[s & 1];
    bf16_t* hAnext = hcp[(s + 1) & 1];
    lstm_cell_kernel<<<dim3(H_ / 16, B_ / 128), 128, 0, stream>>>(
        ZX0, nullptr, hAprev, Wchh0b, H_, nullptr, nullptr, 0, cCondA, hAnext, B_);
    lstm_cell_kernel<<<dim3(H_ / 16, B_ / 128), 128, 0, stream>>>(
        nullptr, cb1, hAnext, Wcih1b, H_,
        condH + (size_t)s * B_ * H_, Wchh1b, H_,
        cCondB, condH + (size_t)(s + 1) * B_ * H_, B_);
  }

  // --- embedding: emb = tanh(cond_h @ eW^T + eb), all NE*B rows at once, bf16 out ---
  gemm_bias_kernel<<<dim3(L_ / 16, MD_ / 128), 128, 0, stream>>>(
      condH + (size_t)B_ * H_, eWb, eb, nullptr, embb, MD_, L_, H_, 1);

  // --- step-invariant decoder input projection: ZE = emb @ dW_ih0[:,61:]^T + db0 ---
  gemm_bias_kernel<<<dim3(G_ / 16, MD_ / 128), 128, 0, stream>>>(
      embb, Web, db0, ZE, nullptr, MD_, G_, L_, 0);

  // --- decoder inputs + init states ---
  build_xtgt_kernel<<<blocks((long long)SL_ * MD_ * KT_), TB, 0, stream>>>(target, Xt);
  init_dec_state_kernel<<<blocks((long long)NE_ * B_ * H_), TB, 0, stream>>>(h0dec, hA0, hB0, cA, cB);

  // --- decoder: 16 sequential steps, all 16 chunks batched (M = 4096) ---
  bf16_t* hAp[2] = {hA0, hA1};
  bf16_t* hBp[2] = {hB0, hB1};
  for (int t = 0; t < SL_; ++t) {
    bf16_t* hAprev = hAp[t & 1];
    bf16_t* hAnext = hAp[(t + 1) & 1];
    bf16_t* hBprev = hBp[t & 1];
    bf16_t* hBnext = hBp[(t + 1) & 1];
    // layer 0: zpre = ZE (emb part + bias), + target slice (K=64) + recurrent (K=1024)
    lstm_cell_kernel<<<dim3(H_ / 16, MD_ / 128), 128, 0, stream>>>(
        ZE, nullptr, Xt + (size_t)t * MD_ * KT_, Wtb, KT_,
        hAprev, Wdhh0b, H_, cA, hAnext, MD_);
    // layer 1
    lstm_cell_kernel<<<dim3(H_ / 16, MD_ / 128), 128, 0, stream>>>(
        nullptr, db1, hAnext, Wdih1b, H_, hBprev, Wdhh1b, H_, cB, hBnext, MD_);
    // output projection + softmax
    gemm_bias_kernel<<<dim3(INP_ / 16, MD_ / 128), 128, 0, stream>>>(
        hBnext, oWb, obp, logits, nullptr, MD_, INP_, H_, 0);
    softmax_out_kernel<<<dim3(MD_ / TB), TB, 0, stream>>>(logits, (float*)d_out, t);
  }
}
